// graph_block_34291018891530
// MI455X (gfx1250) — matrix operand order repaired and checked by reading, not compiled, not run on hardware
//
#include <hip/hip_runtime.h>

// ---------------------------------------------------------------------------
// Graph transformer block for MI455X (gfx1250, wave32, WMMA bf16 16x16x32).
//   1) msg MLP L1 (gathered concat, K=768, LeakyReLU)      -> t1     [E,256] f32
//   2) msg MLP L2                                          -> h      [E,256] f32
//   3) qkv = h @ in_proj_w^T + b   (epilogue stores bf16)  -> qkv_bf [E,768] bf16
//   4) flash attention: LDS-staged K/V chunks, bf16 WMMA   -> o      [E,256] f32
//   5) msg = o @ out_w^T + out_b                           -> msg    [E,256]
//   6) pre = x @ root + bias_p                             -> pre    [N,256]
//   7) scatter-add msg into pre by target node (f32 atomics)
//   8) y = LN1(pre);  9) z = y @ lin_w^T + lin_b;  10) out = LN2(y + z)
// ---------------------------------------------------------------------------

#define D_MODEL 256
#define N_NODES 2048
#define N_EDGES 4096
#define HEADS   8
#define DHEAD   32
#define NEGV    (-10000.0f)

typedef __attribute__((ext_vector_type(16))) __bf16 v16bf;
typedef __attribute__((ext_vector_type(8)))  __bf16 v8bf;
typedef __attribute__((ext_vector_type(8)))  float  v8f;

static __device__ __forceinline__ __bf16 bf(float f) { return (__bf16)f; }

static __device__ __forceinline__ v16bf cat8(v8bf lo, v8bf hi) {
  return __builtin_shufflevector(lo, hi, 0, 1, 2, 3, 4, 5, 6, 7,
                                 8, 9, 10, 11, 12, 13, 14, 15);
}

static __device__ __forceinline__ v8f wmma_bf16(v16bf a, v16bf b, v8f c) {
  // D = A(16x32 bf16) * B(32x16 bf16) + C(16x16 f32)
  return __builtin_amdgcn_wmma_f32_16x16x32_bf16(false, a, false, b,
                                                 (short)0, c, false, false);
}

// ---------------------------------------------------------------------------
// Generic GEMM: out[m][n] = act( sum_k A[m][k] * Wt[k][n] + bias[n] )
//   TRANSW=1: Wt[k][n] = W[n*ldw + k]   (PyTorch Linear weight, x @ W^T)
//   TRANSW=0: Wt[k][n] = W[k*ldw + n]   (plain x @ W)
//   OUTBF16=1: store result as bf16 (for the QKV buffer feeding WMMA).
// One wave per 16x16 output tile; K stepped by 32 through bf16 WMMA.
// ---------------------------------------------------------------------------
template <int TRANSW, int LEAKY, int OUTBF16>
__global__ __launch_bounds__(256) void gemm_bias_kernel(
    const float* __restrict__ A, const float* __restrict__ W,
    const float* __restrict__ bias, float* __restrict__ out,
    int M, int N, int K, int ldw) {
  const int wid  = (blockIdx.x * blockDim.x + threadIdx.x) >> 5;
  const int lane = threadIdx.x & 31;
  const int half = lane >> 4;
  const int lr   = lane & 15;
  const int ntn  = N >> 4;
  if (wid >= (M >> 4) * ntn) return;
  const int m0 = (wid / ntn) << 4;
  const int n0 = (wid % ntn) << 4;

  v8f acc = {0.f, 0.f, 0.f, 0.f, 0.f, 0.f, 0.f, 0.f};
  const float* arow = A + (size_t)(m0 + lr) * K;
  for (int kk = 0; kk < K; kk += 32) {
    v16bf a, b;
#pragma unroll
    for (int i = 0; i < 16; ++i) {
      const int ak = kk + i + 16 * half;                      // A layout
      a[i] = bf(arow[ak]);
      const int bk = kk + i + 16 * half;                      // B layout
      b[i] = bf(TRANSW ? W[(size_t)(n0 + lr) * ldw + bk]
                       : W[(size_t)bk * ldw + (n0 + lr)]);
    }
    acc = wmma_bf16(a, b, acc);
  }
  const float bi = bias[n0 + lr];
#pragma unroll
  for (int r = 0; r < 8; ++r) {
    float v = acc[r] + bi;
    if (LEAKY) v = (v > 0.f) ? v : 0.2f * v;
    const size_t idx = (size_t)(m0 + r + 8 * half) * N + (n0 + lr);  // C layout
    if (OUTBF16)
      reinterpret_cast<__bf16*>(out)[idx] = bf(v);
    else
      out[idx] = v;
  }
}

// ---------------------------------------------------------------------------
// Message MLP layer 1 with fused gather of concat([x_i, x_j, edge_attr]).
// out = LeakyReLU_{0.2}( concat @ W1^T + b1 ),  K = 768.
// ---------------------------------------------------------------------------
__global__ __launch_bounds__(256) void msg_mlp1_kernel(
    const float* __restrict__ x, const int* __restrict__ edge_index,
    const float* __restrict__ edge_attr, const float* __restrict__ W1,
    const float* __restrict__ b1, float* __restrict__ out) {
  const int wid  = (blockIdx.x * blockDim.x + threadIdx.x) >> 5;
  const int lane = threadIdx.x & 31;
  const int half = lane >> 4;
  const int lr   = lane & 15;
  const int ntn  = D_MODEL >> 4;  // 16
  if (wid >= (N_EDGES >> 4) * ntn) return;
  const int m0 = (wid / ntn) << 4;
  const int n0 = (wid % ntn) << 4;

  const int e   = m0 + lr;
  const int src = edge_index[e];             // x_j = x[edge_index[0]]
  const int tgt = edge_index[N_EDGES + e];   // x_i = x[edge_index[1]]
  const float* xi = x + (size_t)tgt * D_MODEL;
  const float* xj = x + (size_t)src * D_MODEL;
  const float* ea = edge_attr + (size_t)e * D_MODEL;

  v8f acc = {0.f, 0.f, 0.f, 0.f, 0.f, 0.f, 0.f, 0.f};
  for (int kk = 0; kk < 3 * D_MODEL; kk += 32) {
    v16bf a, b;
#pragma unroll
    for (int i = 0; i < 16; ++i) {
      const int ak = kk + i + 16 * half;
      float av;
      if (ak < D_MODEL)          av = xi[ak];
      else if (ak < 2 * D_MODEL) av = xj[ak - D_MODEL];
      else                       av = ea[ak - 2 * D_MODEL];
      a[i] = bf(av);
      const int bk = kk + i + 16 * half;
      b[i] = bf(W1[(size_t)(n0 + lr) * (3 * D_MODEL) + bk]);
    }
    acc = wmma_bf16(a, b, acc);
  }
  const float bi = b1[n0 + lr];
#pragma unroll
  for (int r = 0; r < 8; ++r) {
    float v = acc[r] + bi;
    v = (v > 0.f) ? v : 0.2f * v;
    out[(size_t)(m0 + r + 8 * half) * D_MODEL + (n0 + lr)] = v;
  }
}

// ---------------------------------------------------------------------------
// Flash attention over edges. One wave per (head, 16-query tile); the 8 waves
// of a block share one head and 8 consecutive q-tiles, so each 32-key K/V
// chunk is staged ONCE per block into LDS (K key-major, V transposed to
// dh-major) and consumed by all waves as aligned ds_load_b128 fragments.
// Transposed math so query == lane everywhere:
//   S^T = K * Q^T   (A = K tile [16key x 32dh],  B = Q^T [32dh x 16q])
//   O^T = V^T * P^T (A = V^T  [16dh x 32key],   B = P^T  [32key x 16q])
// Online softmax state is one scalar per lane (query), joined across halves
// with a single shfl_xor(16); P^T is rebuilt with cross-half shuffles only.
// ---------------------------------------------------------------------------
__global__ __launch_bounds__(256) void attn_kernel(
    const __bf16* __restrict__ qkv, const unsigned char* __restrict__ mask,
    float* __restrict__ o) {
  __shared__ __bf16 lds_k[32 * 32];  // [key][dh]
  __shared__ __bf16 lds_v[32 * 32];  // [dh][key]  (transposed at staging)

  const int tid  = threadIdx.x;
  const int wid  = blockIdx.x * (blockDim.x >> 5) + (tid >> 5);
  const int lane = tid & 31;
  const int half = lane >> 4;
  const int lr   = lane & 15;
  const int head = wid >> 8;          // E/16 = 256 q-tiles per head
  const int q0   = (wid & 255) << 4;

  const int   LD    = 3 * D_MODEL;
  const float scale = 0.1767766952966369f;  // 1/sqrt(32)

  // Q^T as B fragment: element i -> dh = 8*half + i (i<8), 16 + 8*half + (i-8) (i>=8), column = query = lr.
  v16bf qb;
  {
    const __bf16* qp = qkv + (size_t)(q0 + lr) * LD + head * DHEAD + 8 * half;
    qb = cat8(*(const v8bf*)qp, *(const v8bf*)(qp + 16));
  }

  // Staging role for this thread (same every chunk): 128 threads on K,
  // 128 on V; each moves 8 contiguous dh of one key (16B global load).
  const int st_key = (tid & 127) >> 2;        // 0..31
  const int st_dhb = (tid & 3) << 3;          // 0,8,16,24
  const bool st_is_v = tid >= 128;

  float mstate = -1e30f, lstate = 0.f;
  v8f O0 = {0.f, 0.f, 0.f, 0.f, 0.f, 0.f, 0.f, 0.f};  // dh 0..15 rows
  v8f O1 = {0.f, 0.f, 0.f, 0.f, 0.f, 0.f, 0.f, 0.f};  // dh 16..31 rows

  for (int c0 = 0; c0 < N_EDGES; c0 += 32) {
    __syncthreads();  // previous chunk's LDS reads complete
    {
      const __bf16* gp = qkv + (size_t)(c0 + st_key) * LD +
                         (st_is_v ? 2 * D_MODEL : D_MODEL) + head * DHEAD + st_dhb;
      const v8bf g = *(const v8bf*)gp;
      if (st_is_v) {
#pragma unroll
        for (int j = 0; j < 8; ++j) lds_v[(st_dhb + j) * 32 + st_key] = g[j];
      } else {
        *(v8bf*)&lds_k[st_key * 32 + st_dhb] = g;
      }
    }
    __syncthreads();  // staged chunk visible

    // K tiles as A fragments from LDS: row m = key = lr, k = dh.
    v16bf ka0, ka1;
    {
      const __bf16* k0 = &lds_k[lr * 32 + 8 * half];
      const __bf16* k1 = &lds_k[(16 + lr) * 32 + 8 * half];
      ka0 = cat8(*(const v8bf*)k0, *(const v8bf*)(k0 + 16));
      ka1 = cat8(*(const v8bf*)k1, *(const v8bf*)(k1 + 16));
    }
    v8f S0 = {0.f, 0.f, 0.f, 0.f, 0.f, 0.f, 0.f, 0.f};
    v8f S1 = {0.f, 0.f, 0.f, 0.f, 0.f, 0.f, 0.f, 0.f};
    S0 = wmma_bf16(ka0, qb, S0);  // S^T[key=r+8*half][q=lr], keys c0+0..15
    S1 = wmma_bf16(ka1, qb, S1);  // keys c0+16..31

    // Additive mask: edge_masks[q][key] (bool bytes, 8 consecutive keys).
    const unsigned char* mrow =
        mask + (size_t)(q0 + lr) * N_EDGES + c0 + 8 * half;
    const unsigned long long mb0 = *(const unsigned long long*)(mrow);
    const unsigned long long mb1 = *(const unsigned long long*)(mrow + 16);
    __builtin_prefetch(mrow + 32, 0, 1);  // next chunk's mask row

    float s0[8], s1[8];
    float smax = -1e30f;
#pragma unroll
    for (int r = 0; r < 8; ++r) {
      s0[r] = S0[r] * scale + (((mb0 >> (8 * r)) & 1ull) ? 0.f : NEGV);
      s1[r] = S1[r] * scale + (((mb1 >> (8 * r)) & 1ull) ? 0.f : NEGV);
      smax = fmaxf(smax, fmaxf(s0[r], s1[r]));
    }
    smax = fmaxf(smax, __shfl_xor(smax, 16, 32));  // join halves (same query)
    const float mnew = fmaxf(mstate, smax);
    const float corr = __expf(mstate - mnew);
    float p0[8], p1[8], lsum = 0.f;
#pragma unroll
    for (int r = 0; r < 8; ++r) {
      p0[r] = __expf(s0[r] - mnew);
      p1[r] = __expf(s1[r] - mnew);
      lsum += p0[r] + p1[r];
    }
    lsum  += __shfl_xor(lsum, 16, 32);  // halves hold disjoint keys
    lstate = lstate * corr + lsum;
    mstate = mnew;
#pragma unroll
    for (int r = 0; r < 8; ++r) { O0[r] *= corr; O1[r] *= corr; }

    // P^T as B fragment: element i -> key = i + 16*half, column = query = lr.
    //   half0 needs keys 0..15  : own p0[r] (keys r) + shfl16(p0) (keys 8+r)
    //   half1 needs keys 16..31 : shfl16(p1) (keys 16+r) + own p1[r] (24+r)
    v16bf pb;
#pragma unroll
    for (int r = 0; r < 8; ++r) {
      const float x0 = __shfl_xor(p0[r], 16, 32);
      const float x1 = __shfl_xor(p1[r], 16, 32);
      pb[r]     = bf(half ? x1 : p0[r]);
      pb[r + 8] = bf(half ? p1[r] : x0);
    }

    // V^T tiles as A fragments from LDS (contiguous thanks to transposed
    // staging): row m = dh = lr (+16 for tile1), k = key.
    v16bf va0, va1;
    {
      const __bf16* v0 = &lds_v[lr * 32 + 16 * half];
      const __bf16* v1 = &lds_v[(16 + lr) * 32 + 16 * half];
      va0 = cat8(*(const v8bf*)v0, *(const v8bf*)(v0 + 8));
      va1 = cat8(*(const v8bf*)v1, *(const v8bf*)(v1 + 8));
    }
    O0 = wmma_bf16(va0, pb, O0);
    O1 = wmma_bf16(va1, pb, O1);
  }

  const float inv = 1.f / lstate;
  float* op = o + (size_t)(q0 + lr) * D_MODEL + head * DHEAD;
#pragma unroll
  for (int r = 0; r < 8; ++r) {
    op[r + 8 * half]      = O0[r] * inv;  // O^T row = dh = r+8*half
    op[16 + r + 8 * half] = O1[r] * inv;
  }
}

// ---------------------------------------------------------------------------
// Scatter-add of edge messages into target nodes (global f32 atomics).
// ---------------------------------------------------------------------------
__global__ __launch_bounds__(256) void scatter_add_kernel(
    const float* __restrict__ msg, const int* __restrict__ edge_index,
    float* __restrict__ pre) {
  const int idx = blockIdx.x * blockDim.x + threadIdx.x;
  if (idx >= N_EDGES * D_MODEL) return;
  const int e = idx >> 8;
  const int d = idx & (D_MODEL - 1);
  const int tgt = edge_index[N_EDGES + e];
  atomicAdd(pre + (size_t)tgt * D_MODEL + d, msg[idx]);
}

// ---------------------------------------------------------------------------
// LayerNorm over D=256, one wave per row; optional residual add (in + add).
// ---------------------------------------------------------------------------
__global__ __launch_bounds__(256) void ln_kernel(
    const float* __restrict__ in, const float* __restrict__ add,
    const float* __restrict__ g, const float* __restrict__ b,
    float* __restrict__ out, int rows) {
  const int row  = (blockIdx.x * blockDim.x + threadIdx.x) >> 5;
  const int lane = threadIdx.x & 31;
  if (row >= rows) return;
  const float* ip = in + (size_t)row * D_MODEL + lane * 8;
  const float* ap = add ? add + (size_t)row * D_MODEL + lane * 8 : nullptr;

  float v[8];
  float s = 0.f;
#pragma unroll
  for (int i = 0; i < 8; ++i) {
    float t = ip[i];
    if (ap) t += ap[i];
    v[i] = t;
    s += t;
  }
#pragma unroll
  for (int off = 1; off < 32; off <<= 1) s += __shfl_xor(s, off, 32);
  const float mu = s * (1.f / D_MODEL);
  float q = 0.f;
#pragma unroll
  for (int i = 0; i < 8; ++i) {
    const float d = v[i] - mu;
    q += d * d;
  }
#pragma unroll
  for (int off = 1; off < 32; off <<= 1) q += __shfl_xor(q, off, 32);
  const float rinv = rsqrtf(q * (1.f / D_MODEL) + 1e-5f);
  float* op = out + (size_t)row * D_MODEL + lane * 8;
#pragma unroll
  for (int i = 0; i < 8; ++i)
    op[i] = (v[i] - mu) * rinv * g[lane * 8 + i] + b[lane * 8 + i];
}

// ---------------------------------------------------------------------------
extern "C" void kernel_launch(void* const* d_in, const int* in_sizes, int n_in,
                              void* d_out, int out_size, void* d_ws,
                              size_t ws_size, hipStream_t stream) {
  (void)in_sizes; (void)n_in; (void)out_size; (void)ws_size;
  const float* x          = (const float*)d_in[0];
  const int*   edge_index = (const int*)d_in[1];
  const float* edge_attr  = (const float*)d_in[2];
  const unsigned char* edge_masks = (const unsigned char*)d_in[3];
  const float* W1        = (const float*)d_in[4];
  const float* b1        = (const float*)d_in[5];
  const float* W2        = (const float*)d_in[6];
  const float* b2        = (const float*)d_in[7];
  const float* in_proj_w = (const float*)d_in[8];
  const float* in_proj_b = (const float*)d_in[9];
  const float* out_w     = (const float*)d_in[10];
  const float* out_b     = (const float*)d_in[11];
  const float* root      = (const float*)d_in[12];
  const float* bias_p    = (const float*)d_in[13];
  const float* ln1_g     = (const float*)d_in[14];
  const float* ln1_b     = (const float*)d_in[15];
  const float* ln2_g     = (const float*)d_in[16];
  const float* ln2_b     = (const float*)d_in[17];
  const float* lin_w     = (const float*)d_in[18];
  const float* lin_b     = (const float*)d_in[19];
  float* outp = (float*)d_out;

  // Workspace carve-up (~30 MB).
  float*  t1     = (float*)d_ws;                         // [E, 256] f32
  float*  h      = t1 + (size_t)N_EDGES * D_MODEL;       // [E, 256] f32
  __bf16* qkv_bf = (__bf16*)(h + (size_t)N_EDGES * D_MODEL);  // [E, 768] bf16
  float*  o      = (float*)(qkv_bf + (size_t)N_EDGES * 3 * D_MODEL);  // [E,256]
  float*  msg    = o   + (size_t)N_EDGES * D_MODEL;      // [E, 256]
  float*  pre    = msg + (size_t)N_EDGES * D_MODEL;      // [N, 256]
  float*  y      = pre + (size_t)N_NODES * D_MODEL;      // [N, 256]
  float*  z      = y   + (size_t)N_NODES * D_MODEL;      // [N, 256]

  const int TPB = 256;  // 8 wave32s per block

  // 1) message MLP layer 1 (gather + GEMM K=768 + LeakyReLU)
  msg_mlp1_kernel<<<512, TPB, 0, stream>>>(x, edge_index, edge_attr, W1, b1, t1);
  // 2) message MLP layer 2
  gemm_bias_kernel<1, 0, 0><<<512, TPB, 0, stream>>>(t1, W2, b2, h,
                                                     N_EDGES, D_MODEL, D_MODEL, D_MODEL);
  // 3) QKV projection, stored bf16 for the attention WMMAs
  gemm_bias_kernel<1, 0, 1><<<1536, TPB, 0, stream>>>(h, in_proj_w, in_proj_b,
                                                      (float*)qkv_bf,
                                                      N_EDGES, 3 * D_MODEL, D_MODEL, D_MODEL);
  // 4) flash attention (8 heads x 256 q-tiles = 2048 waves, LDS-staged K/V)
  attn_kernel<<<256, TPB, 0, stream>>>(qkv_bf, edge_masks, o);
  // 5) output projection
  gemm_bias_kernel<1, 0, 0><<<512, TPB, 0, stream>>>(o, out_w, out_b, msg,
                                                     N_EDGES, D_MODEL, D_MODEL, D_MODEL);
  // 6) root transform: pre = x @ root + bias_p (no transpose)
  gemm_bias_kernel<0, 0, 0><<<256, TPB, 0, stream>>>(x, root, bias_p, pre,
                                                     N_NODES, D_MODEL, D_MODEL, D_MODEL);
  // 7) scatter-add messages into target nodes
  scatter_add_kernel<<<(N_EDGES * D_MODEL) / TPB, TPB, 0, stream>>>(msg, edge_index, pre);
  // 8) y = LN1(pre)
  ln_kernel<<<N_NODES / 8, TPB, 0, stream>>>(pre, nullptr, ln1_g, ln1_b, y, N_NODES);
  // 9) z = y @ lin_w^T + lin_b
  gemm_bias_kernel<1, 0, 0><<<256, TPB, 0, stream>>>(y, lin_w, lin_b, z,
                                                     N_NODES, D_MODEL, D_MODEL, D_MODEL);
  // 10) out = LN2(y + z)
  ln_kernel<<<N_NODES / 8, TPB, 0, stream>>>(z, y, ln2_g, ln2_b, outp, N_NODES);
}
